// DyAtGNN_60670708023705
// MI455X (gfx1250) — matrix operand order repaired and checked by reading, not compiled, not run on hardware
//
#include <hip/hip_runtime.h>
#include <hip/hip_bf16.h>
#include <math.h>

// ---- WMMA types (CDNA5 / gfx1250, wave32) ----
typedef __attribute__((ext_vector_type(16))) __bf16          v16bf;
typedef __attribute__((ext_vector_type(16))) unsigned short  v16us;
typedef __attribute__((ext_vector_type(8)))  float           v8f;

union BF16x16 { v16us u; v16bf b; };

__device__ __forceinline__ unsigned short f2bf(float f) {
  unsigned u = __float_as_uint(f);
  u += 0x7FFFu + ((u >> 16) & 1u);       // round-to-nearest-even
  return (unsigned short)(u >> 16);
}

// Fill LDS with the 128x128 f32 weight W converted to bf16, pre-packed in
// B-fragment order: frag (kc,nt), lane l holds col = nt*16+(l&15),
// K = kc*32 + (l>>4)*8 + (j&7) + (j>>3)*16  (j = 0..15)  -> 32 contiguous bytes per lane.
__device__ __forceinline__ void fill_ldsB(v16bf* ldsB, const float* __restrict__ W) {
  for (int idx = threadIdx.x; idx < 4 * 8 * 32; idx += blockDim.x) {
    int kc   = idx >> 8;
    int nt   = (idx >> 5) & 7;
    int lane = idx & 31;
    int col  = nt * 16 + (lane & 15);
    int kb   = kc * 32 + (lane >> 4) * 8;
    BF16x16 t;
#pragma unroll
    for (int j = 0; j < 16; ++j) t.u[j] = f2bf(W[(kb + (j & 7) + ((j >> 3) << 4)) * 128 + col]);
    ldsB[idx] = t.b;
  }
}

// A-fragment (16-bit A 16x32 layout): lane holds row = lane&15; K runs
// {kc*32 + khalf*8 + 0..7} and {kc*32 + 16 + khalf*8 + 0..7}.
__device__ __forceinline__ v16bf make_afrag(const float* __restrict__ arow, int kc, int khalf) {
  int kb = kc * 32 + khalf * 8;
  BF16x16 t;
#pragma unroll
  for (int j = 0; j < 8; ++j) t.u[j]     = f2bf(arow[kb + j]);
#pragma unroll
  for (int j = 0; j < 8; ++j) t.u[8 + j] = f2bf(arow[kb + 16 + j]);
  return t.b;
}

__device__ __forceinline__ v16bf make_afrag_mix(const float* __restrict__ h,
                                                const float* __restrict__ x,
                                                int kc, int khalf) {
  int kb = kc * 32 + khalf * 8;
  BF16x16 t;
#pragma unroll
  for (int j = 0; j < 8; ++j) t.u[j]     = f2bf(0.9f * h[kb + j]      + 0.1f * x[kb + j]);
#pragma unroll
  for (int j = 0; j < 8; ++j) t.u[8 + j] = f2bf(0.9f * h[kb + 16 + j] + 0.1f * x[kb + 16 + j]);
  return t.b;
}

#define WMMA_BF16(A, B, C) \
  __builtin_amdgcn_wmma_f32_16x16x32_bf16(false, (A), false, (B), (short)0, (C), false, false)

// ---------------- Kernel 1: H = relu(A @ W + b),  A:[N,128], W:[128,128] ----
__global__ void __launch_bounds__(256)
k_gemm_bias_relu(const float* __restrict__ A, const float* __restrict__ W,
                 const float* __restrict__ bias, float* __restrict__ C, int nrows) {
  __shared__ v16bf ldsB[4 * 8 * 32];
  fill_ldsB(ldsB, W);
  __syncthreads();

  int wave = threadIdx.x >> 5, lane = threadIdx.x & 31;
  int khalf = lane >> 4, colg = lane & 15;
  int rowbase = blockIdx.x * 128 + wave * 16;
  int arow_i = rowbase + (lane & 15);
  if (arow_i >= nrows) arow_i = nrows - 1;
  const float* arow = A + (size_t)arow_i * 128;

  v8f acc[8];
#pragma unroll
  for (int nt = 0; nt < 8; ++nt)
#pragma unroll
    for (int v = 0; v < 8; ++v) acc[nt][v] = 0.f;

#pragma unroll
  for (int kc = 0; kc < 4; ++kc) {
    v16bf af = make_afrag(arow, kc, khalf);
#pragma unroll
    for (int nt = 0; nt < 8; ++nt)
      acc[nt] = WMMA_BF16(af, ldsB[((kc << 3) + nt) * 32 + lane], acc[nt]);
  }

#pragma unroll
  for (int nt = 0; nt < 8; ++nt) {
    int col = nt * 16 + colg;
    float bv = bias[col];
#pragma unroll
    for (int v = 0; v < 8; ++v) {
      int r = rowbase + khalf * 8 + v;
      if (r < nrows) C[(size_t)r * 128 + col] = fmaxf(acc[nt][v] + bv, 0.f);
    }
  }
}

// ---------------- Kernel 2: exact top-128 (descending, ties -> lower idx) ---
__global__ void __launch_bounds__(1024)
k_topk(const float* __restrict__ scores, const int* __restrict__ remain,
       int nr, int k, int* __restrict__ topk) {
  __shared__ unsigned mask[2816];       // up to 90112 elements
  __shared__ float rv[1024];
  __shared__ int   ri[1024];
  for (int i = threadIdx.x; i < 2816; i += 1024) mask[i] = 0u;
  __syncthreads();
  for (int p = 0; p < k; ++p) {
    float best = -3.4e38f; int bidx = 0x7fffffff;
    for (int i = threadIdx.x; i < nr; i += 1024) {
      if (mask[i >> 5] & (1u << (i & 31))) continue;
      float v = scores[remain[i]];
      if (v > best || (v == best && i < bidx)) { best = v; bidx = i; }
    }
    rv[threadIdx.x] = best; ri[threadIdx.x] = bidx;
    __syncthreads();
    for (int s = 512; s > 0; s >>= 1) {
      if (threadIdx.x < (unsigned)s) {
        float ov = rv[threadIdx.x + s]; int oi = ri[threadIdx.x + s];
        if (ov > rv[threadIdx.x] || (ov == rv[threadIdx.x] && oi < ri[threadIdx.x])) {
          rv[threadIdx.x] = ov; ri[threadIdx.x] = oi;
        }
      }
      __syncthreads();
    }
    if (threadIdx.x == 0) { topk[p] = ri[0]; mask[ri[0] >> 5] |= (1u << (ri[0] & 31)); }
    __syncthreads();
  }
}

// ---------------- Kernel 3: GRU step on pooled rows; scatter back into H ----
__global__ void __launch_bounds__(128)
k_gru(float* __restrict__ H, const int* __restrict__ remain, const int* __restrict__ topk,
      const float* __restrict__ h0, const float* __restrict__ Wih,
      const float* __restrict__ Whh, const float* __restrict__ bih,
      const float* __restrict__ bhh, float* __restrict__ hnew) {
  __shared__ float xrow[128], hrow[128];
  int b = blockIdx.x, t = threadIdx.x;
  int node = remain[topk[b]];
  xrow[t] = H[(size_t)node * 128 + t];
  hrow[t] = h0[b * 128 + t];
  __syncthreads();
  float gx[3], gh[3];
#pragma unroll
  for (int g = 0; g < 3; ++g) {
    const float* wi = Wih + (size_t)(g * 128 + t) * 128;
    const float* wh = Whh + (size_t)(g * 128 + t) * 128;
    float sx = bih[g * 128 + t], sh = bhh[g * 128 + t];
    for (int kk = 0; kk < 128; ++kk) { sx += xrow[kk] * wi[kk]; sh += hrow[kk] * wh[kk]; }
    gx[g] = sx; gh[g] = sh;
  }
  float r = 1.f / (1.f + expf(-(gx[0] + gh[0])));
  float z = 1.f / (1.f + expf(-(gx[1] + gh[1])));
  float n = tanhf(gx[2] + r * gh[2]);
  float hv = (1.f - z) * n + z * hrow[t];
  hnew[b * 128 + t] = hv;
  H[(size_t)node * 128 + t] = hv;   // each block owns a distinct row
}

// ---- Kernel 4: fused attention projections -> al, ar (no 51MB temps) ------
__global__ void __launch_bounds__(256)
k_attn(const float* __restrict__ H, const float* __restrict__ Wl, const float* __restrict__ bl,
       const float* __restrict__ Wr, const float* __restrict__ br,
       const float* __restrict__ avec, float* __restrict__ al, float* __restrict__ ar,
       int nrows) {
  __shared__ v16bf ldsB[4 * 8 * 32];
  __shared__ float red[8][16][16];
  int wave = threadIdx.x >> 5, lane = threadIdx.x & 31;
  int khalf = lane >> 4, colg = lane & 15;
  int rowbase = blockIdx.x * 128 + wave * 16;
  int arow_i = rowbase + (lane & 15);
  if (arow_i >= nrows) arow_i = nrows - 1;
  const float* arow = H + (size_t)arow_i * 128;

  for (int m = 0; m < 2; ++m) {
    __syncthreads();
    fill_ldsB(ldsB, m ? Wr : Wl);
    __syncthreads();
    const float* bias = m ? br : bl;

    v8f acc[8];
#pragma unroll
    for (int nt = 0; nt < 8; ++nt)
#pragma unroll
      for (int v = 0; v < 8; ++v) acc[nt][v] = 0.f;

#pragma unroll
    for (int kc = 0; kc < 4; ++kc) {
      v16bf af = make_afrag(arow, kc, khalf);
#pragma unroll
      for (int nt = 0; nt < 8; ++nt)
        acc[nt] = WMMA_BF16(af, ldsB[((kc << 3) + nt) * 32 + lane], acc[nt]);
    }

    float part[8];
#pragma unroll
    for (int v = 0; v < 8; ++v) part[v] = 0.f;
#pragma unroll
    for (int nt = 0; nt < 8; ++nt) {
      int col = nt * 16 + colg;
      float bv = bias[col], av = avec[col];
#pragma unroll
      for (int v = 0; v < 8; ++v) {
        float y = fmaxf(acc[nt][v] + bv, 0.f);   // relu
        y = (y >= 0.f) ? y : 0.2f * y;           // leaky (identity on relu out)
        part[v] += y * av;
      }
    }
#pragma unroll
    for (int v = 0; v < 8; ++v) red[wave][khalf * 8 + v][colg] = part[v];
    __syncthreads();
    if (lane < 16) {
      float s = 0.f;
#pragma unroll
      for (int c = 0; c < 16; ++c) s += red[wave][lane][c];
      int r = rowbase + lane;
      if (r < nrows) (m ? ar : al)[r] = s;
    }
  }
}

// ---------------- Kernel 5/6: edge softmax pieces --------------------------
__device__ __forceinline__ unsigned fkey(float f) {
  unsigned u = __float_as_uint(f);
  return (u & 0x80000000u) ? ~u : (u | 0x80000000u);
}
__device__ __forceinline__ float funkey(unsigned k) {
  unsigned u = (k & 0x80000000u) ? (k & 0x7FFFFFFFu) : ~k;
  return __uint_as_float(u);
}

__global__ void __launch_bounds__(256)
k_edge_max(const float* __restrict__ al, const float* __restrict__ ar,
           const int* __restrict__ src, const int* __restrict__ dst,
           int E, unsigned* __restrict__ emax) {
  __shared__ unsigned red[256];
  unsigned best = 0u;
  for (int i = blockIdx.x * blockDim.x + threadIdx.x; i < E; i += gridDim.x * blockDim.x) {
    unsigned k = fkey(al[src[i]] + ar[dst[i]]);
    best = best > k ? best : k;
  }
  red[threadIdx.x] = best;
  __syncthreads();
  for (int s = 128; s > 0; s >>= 1) {
    if (threadIdx.x < (unsigned)s)
      red[threadIdx.x] = red[threadIdx.x] > red[threadIdx.x + s] ? red[threadIdx.x] : red[threadIdx.x + s];
    __syncthreads();
  }
  if (threadIdx.x == 0) atomicMax(emax, red[0]);
}

__global__ void __launch_bounds__(256)
k_edge_exp(const float* __restrict__ al, const float* __restrict__ ar,
           const int* __restrict__ src, const int* __restrict__ dst,
           const unsigned* __restrict__ emax, float* __restrict__ e,
           float* __restrict__ rowsum, int E) {
  int i = blockIdx.x * blockDim.x + threadIdx.x;
  if (i >= E) return;
  float mx = funkey(*emax);
  int s = src[i];
  float v = expf(al[s] + ar[dst[i]] - mx);
  e[i] = v;
  atomicAdd(&rowsum[s], v);
}

// ---------------- Kernel 7: SpMM scatter-add (A^T @ feats) -----------------
__global__ void __launch_bounds__(256)
k_spmm(const float* __restrict__ e, const float* __restrict__ rowsum,
       const int* __restrict__ src, const int* __restrict__ dst,
       const float* __restrict__ in, float* __restrict__ hi, int E) {
  long long gid = (long long)blockIdx.x * blockDim.x + threadIdx.x;
  int ed = (int)(gid >> 4);
  if (ed >= E) return;
  int c = (int)(gid & 15);
  int s = src[ed], d = dst[ed];
  float w = e[ed] / (rowsum[s] + 1e-16f);
  const float* ip = in + (size_t)s * 128 + c * 8;
  float*       op = hi + (size_t)d * 128 + c * 8;
  float4 a0 = *(const float4*)(ip);
  float4 a1 = *(const float4*)(ip + 4);
  atomicAdd(op + 0, w * a0.x); atomicAdd(op + 1, w * a0.y);
  atomicAdd(op + 2, w * a0.z); atomicAdd(op + 3, w * a0.w);
  atomicAdd(op + 4, w * a1.x); atomicAdd(op + 5, w * a1.y);
  atomicAdd(op + 6, w * a1.z); atomicAdd(op + 7, w * a1.w);
}

// ---- Kernel 8: out = relu(theta*(S@hnew) + (1-theta)*S), S=0.9*hi+0.1*Xres
__global__ void __launch_bounds__(256)
k_conv(const float* __restrict__ hi, const float* __restrict__ Xres,
       const float* __restrict__ hnew, float theta, float* __restrict__ out, int nrows) {
  __shared__ v16bf ldsB[4 * 8 * 32];
  fill_ldsB(ldsB, hnew);
  __syncthreads();
  int wave = threadIdx.x >> 5, lane = threadIdx.x & 31;
  int khalf = lane >> 4, colg = lane & 15;
  int rowbase = blockIdx.x * 128 + wave * 16;
  int arow_i = rowbase + (lane & 15);
  if (arow_i >= nrows) arow_i = nrows - 1;
  const float* hrow = hi   + (size_t)arow_i * 128;
  const float* xrow = Xres + (size_t)arow_i * 128;

  v8f acc[8];
#pragma unroll
  for (int nt = 0; nt < 8; ++nt)
#pragma unroll
    for (int v = 0; v < 8; ++v) acc[nt][v] = 0.f;

#pragma unroll
  for (int kc = 0; kc < 4; ++kc) {
    v16bf af = make_afrag_mix(hrow, xrow, kc, khalf);
#pragma unroll
    for (int nt = 0; nt < 8; ++nt)
      acc[nt] = WMMA_BF16(af, ldsB[((kc << 3) + nt) * 32 + lane], acc[nt]);
  }

  float omt = 1.f - theta;
#pragma unroll
  for (int nt = 0; nt < 8; ++nt) {
    int col = nt * 16 + colg;
#pragma unroll
    for (int v = 0; v < 8; ++v) {
      int r = rowbase + khalf * 8 + v;
      if (r < nrows) {
        size_t o = (size_t)r * 128 + col;
        float sval = 0.9f * hi[o] + 0.1f * Xres[o];
        out[o] = fmaxf(theta * acc[nt][v] + omt * sval, 0.f);
      }
    }
  }
}

// ---------------------------------------------------------------------------
extern "C" void kernel_launch(void* const* d_in, const int* in_sizes, int n_in,
                              void* d_out, int out_size, void* d_ws, size_t ws_size,
                              hipStream_t stream) {
  const float* x      = (const float*)d_in[0];
  const float* scores = (const float*)d_in[1];
  const float* W_hid  = (const float*)d_in[2];
  const float* b_hid  = (const float*)d_in[3];
  const float* Wl     = (const float*)d_in[4];
  const float* bl     = (const float*)d_in[5];
  const float* Wr     = (const float*)d_in[6];
  const float* br     = (const float*)d_in[7];
  const float* avec   = (const float*)d_in[8];
  const float* h0w    = (const float*)d_in[9];
  const float* W_ih   = (const float*)d_in[10];
  const float* W_hh   = (const float*)d_in[11];
  const float* b_ih   = (const float*)d_in[12];
  const float* b_hh   = (const float*)d_in[13];
  const int*   edge   = (const int*)d_in[14];
  const int*   remain = (const int*)d_in[15];

  int N  = in_sizes[0] / 128;
  int E  = in_sizes[14] / 2;
  int NR = in_sizes[15];
  const int* src = edge;
  const int* dst = edge + E;
  float* out = (float*)d_out;

  // workspace layout (~110 MB)
  float* H      = (float*)d_ws;                   // [N,128] (= X_res = X_gcn)
  float* hi     = H + (size_t)N * 128;            // [N,128]
  float* ebuf   = hi + (size_t)N * 128;           // [E]
  float* rowsum = ebuf + E;                       // [N]
  float* al     = rowsum + N;                     // [N]
  float* ar     = al + N;                         // [N]
  float* hnew   = ar + N;                         // [128,128]
  int*   topk   = (int*)(hnew + 128 * 128);       // [128]
  unsigned* emax = (unsigned*)(topk + 128);       // [1]

  int gb = (N + 127) / 128;

  k_gemm_bias_relu<<<gb, 256, 0, stream>>>(x, W_hid, b_hid, H, N);
  k_topk<<<1, 1024, 0, stream>>>(scores, remain, NR, 128, topk);
  k_gru<<<128, 128, 0, stream>>>(H, remain, topk, h0w, W_ih, W_hh, b_ih, b_hh, hnew);
  k_attn<<<gb, 256, 0, stream>>>(H, Wl, bl, Wr, br, avec, al, ar, N);

  hipMemsetAsync(emax, 0, sizeof(unsigned), stream);
  k_edge_max<<<1024, 256, 0, stream>>>(al, ar, src, dst, E, emax);
  hipMemsetAsync(rowsum, 0, (size_t)N * sizeof(float), stream);
  k_edge_exp<<<(E + 255) / 256, 256, 0, stream>>>(al, ar, src, dst, emax, ebuf, rowsum, E);

  const float* inb = H;
  for (int l = 1; l <= 2; ++l) {
    float theta = logf(0.5f / (float)l + 1.0f);
    hipMemsetAsync(hi, 0, (size_t)N * 128 * sizeof(float), stream);
    long long tt = (long long)E * 16;
    k_spmm<<<(int)((tt + 255) / 256), 256, 0, stream>>>(ebuf, rowsum, src, dst, inb, hi, E);
    k_conv<<<gb, 256, 0, stream>>>(hi, H, hnew, theta, out, N);
    inb = out;
  }
}
